// MorsePotentialChemical_77756087927055
// MI455X (gfx1250) — compile-verified, matrix operand order checked by reading, not run on hardware
//
#include <hip/hip_runtime.h>

#define ALPHA_C   2.8f
#define R_ONSET_C 1.7f
#define R_CUT_C   2.0f
#define NP 4096
#define NC 16
#define NTILE (NP / 16)   // 256 tiles per dimension

typedef _Float16 v8h  __attribute__((ext_vector_type(8)));
typedef _Float16 v16h __attribute__((ext_vector_type(16)));
typedef float    v8f  __attribute__((ext_vector_type(8)));

// ---------------------------------------------------------------------------
// Zero the scalar output (harness poisons d_out with 0xAA before timing).
// ---------------------------------------------------------------------------
__global__ void morse_zero_out(float* __restrict__ out, int n) {
    int i = blockIdx.x * blockDim.x + threadIdx.x;
    if (i < n) out[i] = 0.0f;
}

// ---------------------------------------------------------------------------
// Prep: per particle i build
//   w_i[d]  = sum_c ct[i,c] * 0.5*(E[i,c,d] + E[i,d,c])
//   U_i     = [w_i ; ct_i]   (32 x f16)  -> WMMA A operand rows
//   V_i     = [ct_i ; w_i]   (32 x f16)  -> WMMA B operand rows
//   csum[i] = sum_c ct[i,c]              -> column mask
// One thread per (i,d); N*16 = 65536 threads.
// ---------------------------------------------------------------------------
__global__ __launch_bounds__(256)
void morse_prep_kernel(const float* __restrict__ celltype,
                       const float* __restrict__ epsilon,
                       _Float16* __restrict__ Uh,
                       _Float16* __restrict__ Vh,
                       float* __restrict__ csum) {
    int gid = blockIdx.x * blockDim.x + threadIdx.x;
    int i = gid >> 4;
    int d = gid & 15;
    const float* ct = celltype + i * NC;
    const float* E  = epsilon + i * NC * NC;

    float w = 0.0f;
#pragma unroll
    for (int c = 0; c < NC; ++c)
        w += ct[c] * 0.5f * (E[c * NC + d] + E[d * NC + c]);

    float cd = ct[d];
    Uh[i * 32 + d]      = (_Float16)w;
    Uh[i * 32 + 16 + d] = (_Float16)cd;
    Vh[i * 32 + d]      = (_Float16)cd;
    Vh[i * 32 + 16 + d] = (_Float16)w;

    if (d == 0) {
        float s = 0.0f;
#pragma unroll
        for (int c = 0; c < NC; ++c) s += ct[c];
        csum[i] = s;
    }
}

// ---------------------------------------------------------------------------
// Fused tile kernel: one wave32 owns a 16x32 slab of the pair matrix
// (one i-tile x two adjacent j-tiles). WMMA f32_16x16x32_f16 computes
// eps_raw = U_i . V_j for the whole tile, then the same wave evaluates the
// Morse energy + smooth cutoff directly on the accumulator registers,
// reduces across the wave, and does one atomicAdd. The N x N eps matrix is
// never materialized (saves ~134 MB of HBM traffic vs a two-pass scheme).
//
// 16-bit A operand layout (ISA 7.12.2): lane m=lane&15 holds, ascending in
// the 16 half-registers, K = off..off+7 then K = off+16..off+23 where
// off = 8*(lane>=16). B is the mirrored layout with n=lane&15.
// f32 C/D layout: VGPR v, lane l -> row v + 8*(l>=16), col l&15.
// ---------------------------------------------------------------------------
__global__ __launch_bounds__(256)
void morse_pair_kernel(const float* __restrict__ position,
                       const float* __restrict__ radius,
                       const _Float16* __restrict__ Uh,
                       const _Float16* __restrict__ Vh,
                       const float* __restrict__ csum,
                       float* __restrict__ out) {
    const int lane      = threadIdx.x & 31;
    const int waveInBlk = threadIdx.x >> 5;
    // wave-tile id: each wave covers i-tile `it`, j-tiles {2*jp, 2*jp+1}
    const int wt = blockIdx.x * 8 + waveInBlk;
    const int it = wt / (NTILE / 2);
    const int jp = wt % (NTILE / 2);
    const int ibase  = it * 16;
    const int jbase0 = jp * 32;
    const int jbase1 = jbase0 + 16;

    const int lr  = lane & 15;
    const int hi  = lane >> 4;       // 0 for lanes 0-15, 1 for lanes 16-31
    const int off = hi * 8;          // K-offset of this lane's chunk

    // --- A operand: row (ibase+lr) of U, two contiguous 8-halfword chunks ---
    v16h a;
    {
        const _Float16* up = Uh + (ibase + lr) * 32 + off;
        v8h lo = *(const v8h*)(up);
        v8h hh = *(const v8h*)(up + 16);
#pragma unroll
        for (int k = 0; k < 8; ++k) { a[k] = lo[k]; a[k + 8] = hh[k]; }
    }
    // --- B operands: rows (jbase+lr) of V for the two j-tiles ---
    v16h b0, b1;
    {
        const _Float16* vp = Vh + (jbase0 + lr) * 32 + off;
        v8h lo = *(const v8h*)(vp);
        v8h hh = *(const v8h*)(vp + 16);
#pragma unroll
        for (int k = 0; k < 8; ++k) { b0[k] = lo[k]; b0[k + 8] = hh[k]; }
    }
    {
        const _Float16* vp = Vh + (jbase1 + lr) * 32 + off;
        v8h lo = *(const v8h*)(vp);
        v8h hh = *(const v8h*)(vp + 16);
#pragma unroll
        for (int k = 0; k < 8; ++k) { b1[k] = lo[k]; b1[k + 8] = hh[k]; }
    }

    v8f z = {};
    v8f acc[2];
    // D = A x B + 0 ; K=32 in a single instruction each; A reused back-to-back
    acc[0] = __builtin_amdgcn_wmma_f32_16x16x32_f16(
        false, a, false, b0, (short)0, z, false, false);
    acc[1] = __builtin_amdgcn_wmma_f32_16x16x32_f16(
        false, a, false, b1, (short)0, z, false, false);

    // --- fused Morse energy on the accumulator layout ---
    const float rc2 = R_CUT_C * R_CUT_C;
    const float ro2 = R_ONSET_C * R_ONSET_C;
    const float inv_den = 1.0f / ((rc2 - ro2) * (rc2 - ro2) * (rc2 - ro2));

    float partial = 0.0f;
#pragma unroll
    for (int half = 0; half < 2; ++half) {
        const int jb = half ? jbase1 : jbase0;
        const int j  = jb + lr;                 // this lane's column
        const float pjx = position[3 * j + 0];
        const float pjy = position[3 * j + 1];
        const float pjz = position[3 * j + 2];
        const float rj    = radius[j];
        const float maskj = (csum[j] > 0.0f) ? 1.0f : 0.0f;
#pragma unroll
        for (int v = 0; v < 8; ++v) {
            const int i = ibase + v + (hi << 3); // this element's row
            if (i == j) continue;                // diagonal mask
            const float dx = position[3 * i + 0] - pjx;
            const float dy = position[3 * i + 1] - pjy;
            const float dz = position[3 * i + 2] - pjz;
            const float r2 = dx * dx + dy * dy + dz * dz;
            if (r2 >= rc2) continue;             // cutoff -> e == 0
            // hardware-native transcendentals: v_sqrt_f32 / v_rcp_f32 /
            // v_exp_f32, no precise-div / Newton expansion in the N^2 path
            const float dr = __builtin_amdgcn_sqrtf(r2);
            float cut;
            if (dr < R_ONSET_C) {
                cut = 1.0f;
            } else {
                const float t = rc2 - r2;
                cut = t * t * (rc2 + 2.0f * r2 - 3.0f * ro2) * inv_den;
            }
            const float x   = 0.5f * acc[half][v];            // eps_raw
            const float sigmoid = __builtin_amdgcn_rcpf(1.0f + __expf(-x));
            const float eps = (5.0f * sigmoid + 0.3f) * maskj;
            const float sg  = radius[i] + rj;
            const float u   = 1.0f - __expf(-ALPHA_C * (dr - sg));
            partial += (eps * u * u - eps) * cut;
        }
    }

    // --- wave32 reduction, one atomic per wave ---
#pragma unroll
    for (int o = 16; o > 0; o >>= 1)
        partial += __shfl_xor(partial, o, 32);
    if (lane == 0)
        atomicAdd(out, 0.5f * partial);
}

// ---------------------------------------------------------------------------
extern "C" void kernel_launch(void* const* d_in, const int* in_sizes, int n_in,
                              void* d_out, int out_size, void* d_ws, size_t ws_size,
                              hipStream_t stream) {
    (void)in_sizes; (void)n_in; (void)ws_size;

    const float* position = (const float*)d_in[0];   // (N,3)
    const float* celltype = (const float*)d_in[1];   // (N,16)
    const float* epsilon  = (const float*)d_in[2];   // (N,256)
    const float* radius   = (const float*)d_in[3];   // (N,1)
    float* out = (float*)d_out;

    // workspace: U (N x 32 f16), V (N x 32 f16), csum (N f32) = ~528 KB
    _Float16* Uh = (_Float16*)d_ws;
    _Float16* Vh = Uh + (size_t)NP * 32;
    float* csum  = (float*)(Vh + (size_t)NP * 32);

    morse_zero_out<<<1, 64, 0, stream>>>(out, out_size);

    morse_prep_kernel<<<(NP * NC) / 256, 256, 0, stream>>>(
        celltype, epsilon, Uh, Vh, csum);

    // 256 i-tiles x 128 double-j-tiles = 32768 wave-tiles, 8 waves/block
    morse_pair_kernel<<<32768 / 8, 256, 0, stream>>>(
        position, radius, Uh, Vh, csum, out);
}
